// CausalSelfAttentionMemory_55387898249523
// MI455X (gfx1250) — compile-verified, matrix operand order checked by reading, not run on hardware
//
#include <hip/hip_runtime.h>
#include <hip/hip_bf16.h>

// ---------------------------------------------------------------------------
// Causal self-attention (B=2, C=2048, D=2048, H=16, HD=128), fp32 in/out.
// Strategy: one-shot f32->bf16 conversion of x + weights, then all GEMMs run
// pure bf16 v_wmma_f32_16x16x32_bf16 with 32x64 register tiles per wave.
// Flash attention stages K/V tiles through LDS (8 waves share each tile).
// ---------------------------------------------------------------------------

#define BB  2
#define CC  2048
#define DD  2048
#define HH  16
#define HDIM 128
#define HD2  64          // HDIM/2 (rope freq count)
#define MTOT (BB*CC)     // 4096 rows of x

typedef __attribute__((ext_vector_type(16))) __bf16 v16bf;
typedef __attribute__((ext_vector_type(8)))  __bf16 bf16x8;
typedef __attribute__((ext_vector_type(8)))  float  v8f;
typedef __attribute__((ext_vector_type(4)))  float  f32x4;

// A-fragment (and, by layout symmetry, B-fragment of the transposed matrix)
// for v_wmma_*_16x16x32: lane (m = lane&15, h = lane>>4) holds
//   elements 0..7  : row m, K = k0 + h*8 + {0..7}
//   elements 8..15 : row m, K = k0 + 16 + h*8 + {0..7}
// Both 8-element chunks are 16B-contiguous -> global_load_b128 / ds_load_b128.
__device__ __forceinline__ v16bf load_frag_bf16(const __bf16* base,
                                                int ld, int k0, int lane) {
    const int m  = lane & 15;
    const int hf = (lane >> 4) & 1;
    const __bf16* p0 = base + (size_t)m * ld + k0 + hf * 8;
    bf16x8 lo = *(const bf16x8*)p0;
    bf16x8 hi = *(const bf16x8*)(p0 + 16);
    v16bf f;
#pragma unroll
    for (int j = 0; j < 8; ++j) { f[j] = lo[j]; f[8 + j] = hi[j]; }
    return f;
}

__device__ __forceinline__ v8f wmma_bf16(v16bf a, v16bf b, v8f c) {
    return __builtin_amdgcn_wmma_f32_16x16x32_bf16(
        /*neg_a=*/false, a, /*neg_b=*/false, b,
        /*c_mod=*/(short)0, c, /*reuse_a=*/false, /*reuse_b=*/false);
}

// ---------------------------------------------------------------------------
// Kernel 0: bulk f32 -> bf16 conversion (8 elements / thread, b128 traffic).
// ---------------------------------------------------------------------------
__global__ void cvt_bf16_kernel(const float* __restrict__ src,
                                __bf16* __restrict__ dst, int n8) {
    int i = blockIdx.x * blockDim.x + threadIdx.x;
    if (i >= n8) return;
    f32x4 a = *(const f32x4*)(src + (size_t)i * 8);
    f32x4 b = *(const f32x4*)(src + (size_t)i * 8 + 4);
    bf16x8 o;
#pragma unroll
    for (int j = 0; j < 4; ++j) { o[j] = (__bf16)a[j]; o[4 + j] = (__bf16)b[j]; }
    *(bf16x8*)(dst + (size_t)i * 8) = o;
}

// ---------------------------------------------------------------------------
// Kernel 1: y = x @ W^T for W in {Wq,Wk,Wv} (blockIdx.z selects), RoPE on q,k.
// 32x64 output per wave (2M x 4N register tile, 8 WMMAs / K-step).
// q,k stored bf16 head-major [B][H][C][HD]; v stored transposed [B][H][HD][C].
// grid (MTOT/32, DD/512, 3), block 256 (8 waves along N).
// ---------------------------------------------------------------------------
__global__ void qkv_rope_kernel(const __bf16* __restrict__ xb,
                                const float* __restrict__ fcos,
                                const float* __restrict__ fsin,
                                const __bf16* __restrict__ Wqb,
                                const __bf16* __restrict__ Wkb,
                                const __bf16* __restrict__ Wvb,
                                __bf16* __restrict__ qb,
                                __bf16* __restrict__ kb,
                                __bf16* __restrict__ vT) {
    const int lane = threadIdx.x & 31;
    const int wave = threadIdx.x >> 5;
    const int m0 = blockIdx.x * 32;
    const int n0 = blockIdx.y * 512 + wave * 64;
    const int z  = blockIdx.z;
    const __bf16* W = (z == 0) ? Wqb : ((z == 1) ? Wkb : Wvb);

    v8f acc[2][4];
#pragma unroll
    for (int mt = 0; mt < 2; ++mt)
#pragma unroll
        for (int nt = 0; nt < 4; ++nt)
#pragma unroll
            for (int i = 0; i < 8; ++i) acc[mt][nt][i] = 0.0f;

    const __bf16* xrow0 = xb + (size_t)m0 * DD;
    const __bf16* xrow1 = xb + (size_t)(m0 + 16) * DD;
    for (int k0 = 0; k0 < DD; k0 += 32) {
        v16bf a0 = load_frag_bf16(xrow0, DD, k0, lane);
        v16bf a1 = load_frag_bf16(xrow1, DD, k0, lane);
#pragma unroll
        for (int nt = 0; nt < 4; ++nt) {
            v16bf b = load_frag_bf16(W + (size_t)(n0 + nt * 16) * DD, DD, k0, lane);
            acc[0][nt] = wmma_bf16(a0, b, acc[0][nt]);
            acc[1][nt] = wmma_bf16(a1, b, acc[1][nt]);
        }
    }

    const int hf = (lane >> 4) & 1;
    const int ln = lane & 15;
#pragma unroll
    for (int nt = 0; nt < 4; ++nt) {
        const int n  = n0 + nt * 16 + ln;
        const int hh = n / HDIM;
        const int d  = n % HDIM;
        const int jf = d >> 1;
#pragma unroll
        for (int mt = 0; mt < 2; ++mt) {
#pragma unroll
            for (int r = 0; r < 8; ++r) {
                const int m  = m0 + mt * 16 + r + hf * 8;
                const int bz = m / CC;
                const int c  = m % CC;
                float val = acc[mt][nt][r];
                if (z < 2) {
                    // RoPE: pair (even d, odd d) lives in adjacent lanes.
                    float other = __shfl_xor(val, 1, 32);
                    float cs = fcos[c * HD2 + jf];
                    float sn = fsin[c * HD2 + jf];
                    val = (d & 1) ? (other * sn + val * cs)
                                  : (val * cs - other * sn);
                }
                __bf16 bv = (__bf16)val;
                const size_t head = (size_t)bz * HH + hh;
                if (z == 0)      qb[(head * CC + c) * HDIM + d] = bv;
                else if (z == 1) kb[(head * CC + c) * HDIM + d] = bv;
                else             vT[(head * HDIM + d) * CC + c] = bv;
            }
        }
    }
}

// ---------------------------------------------------------------------------
// Kernel 2: flash attention. Block = 8 waves covering 128 consecutive queries
// of one (b,h); each 32-key K/V tile is staged once into LDS and shared by
// all 8 waves (8x global traffic reduction). grid (CC/128, HH, BB), block 256.
// ---------------------------------------------------------------------------
__global__ void attn_kernel(const __bf16* __restrict__ qb,
                            const __bf16* __restrict__ kb,
                            const __bf16* __restrict__ vT,
                            __bf16* __restrict__ attn) {
    __shared__ __bf16 ldsK[32 * HDIM];      // [key][d]   8 KB
    __shared__ __bf16 ldsV[HDIM * 32];      // [d][key]   8 KB
    __shared__ __bf16 ldsP[8 * 16 * 32];    // per-wave P bounce, 8 KB

    const int tid  = threadIdx.x;
    const int lane = tid & 31;
    const int wave = tid >> 5;
    const int hf   = (lane >> 4) & 1;
    const int ln   = lane & 15;
    const int q0b  = blockIdx.x * 128;      // block's query base
    const int q0   = q0b + wave * 16;       // this wave's query tile
    const size_t head = (size_t)blockIdx.z * HH + blockIdx.y;

    const __bf16* qh = qb + head * CC * HDIM;
    const __bf16* kh = kb + head * CC * HDIM;
    const __bf16* vh = vT + head * HDIM * CC;
    __bf16* myP = &ldsP[wave * 512];

    // staging indices (fixed per thread): K tile 32x128, V tile 128x32
    const int krow = tid >> 3;              // 0..31
    const int kcol = (tid & 7) * 16;        // 0..112
    const int vrow = tid >> 1;              // 0..127
    const int vcol = (tid & 1) * 16;        // 0 or 16

    // Resident Q fragments covering HD=128 (4 K-chunks of 32).
    v16bf qf[4];
#pragma unroll
    for (int t = 0; t < 4; ++t)
        qf[t] = load_frag_bf16(qh + (size_t)q0 * HDIM, HDIM, t * 32, lane);

    v8f acc[8];
#pragma unroll
    for (int t = 0; t < 8; ++t)
#pragma unroll
        for (int i = 0; i < 8; ++i) acc[t][i] = 0.0f;

    float mrow[8], lrow[8];
#pragma unroll
    for (int r = 0; r < 8; ++r) { mrow[r] = -1.0e30f; lrow[r] = 0.0f; }

    const float scale = 0.08838834764831845f;  // 1/sqrt(128)
    const int kend = q0b + 128;                // block-uniform causal bound

    for (int kblk = 0; kblk < kend; kblk += 32) {
        // ---- cooperative staging of K (32x128) and V (128x32) tiles ----
        {
            const __bf16* ksrc = kh + (size_t)(kblk + krow) * HDIM + kcol;
            *(bf16x8*)&ldsK[krow * HDIM + kcol]     = *(const bf16x8*)ksrc;
            *(bf16x8*)&ldsK[krow * HDIM + kcol + 8] = *(const bf16x8*)(ksrc + 8);
            const __bf16* vsrc = vh + (size_t)vrow * CC + kblk + vcol;
            *(bf16x8*)&ldsV[vrow * 32 + vcol]     = *(const bf16x8*)vsrc;
            *(bf16x8*)&ldsV[vrow * 32 + vcol + 8] = *(const bf16x8*)(vsrc + 8);
            if (kblk + 32 < kend) {   // prefetch next tile (global_prefetch_b8)
                __builtin_prefetch(kh + (size_t)(kblk + 32 + krow) * HDIM + kcol, 0, 1);
                __builtin_prefetch(vh + (size_t)vrow * CC + kblk + 32 + vcol, 0, 1);
            }
        }
        __syncthreads();

        // ---- scores: two 16x16 N-tiles (keys kblk..+15, +16..+31) ----
        v8f s0, s1;
#pragma unroll
        for (int i = 0; i < 8; ++i) { s0[i] = 0.0f; s1[i] = 0.0f; }
#pragma unroll
        for (int t = 0; t < 4; ++t) {
            v16bf kf0 = load_frag_bf16(&ldsK[0], HDIM, t * 32, lane);
            s0 = wmma_bf16(qf[t], kf0, s0);
            v16bf kf1 = load_frag_bf16(&ldsK[16 * HDIM], HDIM, t * 32, lane);
            s1 = wmma_bf16(qf[t], kf1, s1);
        }

        // ---- causal mask + online softmax (rows live across 16 lanes) ----
        const int key0 = kblk + ln;
        const int key1 = kblk + 16 + ln;
#pragma unroll
        for (int r = 0; r < 8; ++r) {
            const int qpos = q0 + r + hf * 8;
            float a0 = s0[r] * scale; if (key0 > qpos) a0 = -1.0e30f;
            float a1 = s1[r] * scale; if (key1 > qpos) a1 = -1.0e30f;
            float mx = fmaxf(a0, a1);
            mx = fmaxf(mx, __shfl_xor(mx, 8, 32));
            mx = fmaxf(mx, __shfl_xor(mx, 4, 32));
            mx = fmaxf(mx, __shfl_xor(mx, 2, 32));
            mx = fmaxf(mx, __shfl_xor(mx, 1, 32));
            const float mnew  = fmaxf(mrow[r], mx);
            const float alpha = __expf(mrow[r] - mnew);
            const float p0 = __expf(a0 - mnew);
            const float p1 = __expf(a1 - mnew);
            float sum = p0 + p1;
            sum += __shfl_xor(sum, 8, 32);
            sum += __shfl_xor(sum, 4, 32);
            sum += __shfl_xor(sum, 2, 32);
            sum += __shfl_xor(sum, 1, 32);
            lrow[r] = lrow[r] * alpha + sum;
            mrow[r] = mnew;
#pragma unroll
            for (int t = 0; t < 8; ++t) acc[t][r] *= alpha;
            const int mi = r + hf * 8;
            myP[mi * 32 + ln]      = (__bf16)p0;
            myP[mi * 32 + 16 + ln] = (__bf16)p1;
        }
        __syncthreads();

        // ---- re-read P in A-fragment layout; acc += P @ V from LDS ----
        v16bf pf;
        {
            const __bf16* pp = &myP[ln * 32 + hf * 8];
#pragma unroll
            for (int j = 0; j < 8; ++j) { pf[j] = pp[j]; pf[8 + j] = pp[16 + j]; }
        }
#pragma unroll
        for (int t = 0; t < 8; ++t) {
            v16bf vf = load_frag_bf16(&ldsV[(t * 16) * 32], 32, 0, lane);
            acc[t] = wmma_bf16(pf, vf, acc[t]);
        }
        __syncthreads();   // protect LDS tiles before next staging
    }

    // ---- normalize, emit attn_cat (b, c, h*HD+d) as bf16 ----
#pragma unroll
    for (int r = 0; r < 8; ++r) {
        const float inv = 1.0f / lrow[r];
        const int c = q0 + r + hf * 8;
        const size_t rowbase = ((size_t)blockIdx.z * CC + c) * DD
                             + (size_t)blockIdx.y * HDIM;
#pragma unroll
        for (int t = 0; t < 8; ++t)
            attn[rowbase + t * 16 + ln] = (__bf16)(acc[t][r] * inv);
    }
}

// ---------------------------------------------------------------------------
// Kernel 3: out = attn_cat @ Wo^T (f32 result), 32x64 register tile per wave.
// grid (MTOT/32, DD/512), block 256.
// ---------------------------------------------------------------------------
__global__ void out_proj_kernel(const __bf16* __restrict__ attn,
                                const __bf16* __restrict__ Wob,
                                float* __restrict__ out) {
    const int lane = threadIdx.x & 31;
    const int wave = threadIdx.x >> 5;
    const int m0 = blockIdx.x * 32;
    const int n0 = blockIdx.y * 512 + wave * 64;

    v8f acc[2][4];
#pragma unroll
    for (int mt = 0; mt < 2; ++mt)
#pragma unroll
        for (int nt = 0; nt < 4; ++nt)
#pragma unroll
            for (int i = 0; i < 8; ++i) acc[mt][nt][i] = 0.0f;

    const __bf16* arow0 = attn + (size_t)m0 * DD;
    const __bf16* arow1 = attn + (size_t)(m0 + 16) * DD;
    for (int k0 = 0; k0 < DD; k0 += 32) {
        v16bf a0 = load_frag_bf16(arow0, DD, k0, lane);
        v16bf a1 = load_frag_bf16(arow1, DD, k0, lane);
#pragma unroll
        for (int nt = 0; nt < 4; ++nt) {
            v16bf b = load_frag_bf16(Wob + (size_t)(n0 + nt * 16) * DD, DD, k0, lane);
            acc[0][nt] = wmma_bf16(a0, b, acc[0][nt]);
            acc[1][nt] = wmma_bf16(a1, b, acc[1][nt]);
        }
    }

    const int hf = (lane >> 4) & 1;
    const int ln = lane & 15;
#pragma unroll
    for (int mt = 0; mt < 2; ++mt)
#pragma unroll
        for (int nt = 0; nt < 4; ++nt) {
            const int n = n0 + nt * 16 + ln;
#pragma unroll
            for (int r = 0; r < 8; ++r) {
                const int m = m0 + mt * 16 + r + hf * 8;
                out[(size_t)m * DD + n] = acc[mt][nt][r];
            }
        }
}

// ---------------------------------------------------------------------------
extern "C" void kernel_launch(void* const* d_in, const int* in_sizes, int n_in,
                              void* d_out, int out_size, void* d_ws, size_t ws_size,
                              hipStream_t stream) {
    const float* x    = (const float*)d_in[0];
    const float* fcos = (const float*)d_in[1];
    const float* fsin = (const float*)d_in[2];
    const float* Wq   = (const float*)d_in[3];
    const float* Wk   = (const float*)d_in[4];
    const float* Wv   = (const float*)d_in[5];
    const float* Wo   = (const float*)d_in[6];

    const size_t NE = (size_t)BB * HH * CC * HDIM;  // 8,388,608
    const size_t NW = (size_t)DD * DD;              // 4,194,304
    __bf16* qbuf = (__bf16*)d_ws;
    __bf16* kbuf = qbuf + NE;
    __bf16* vT   = kbuf + NE;
    __bf16* attn = vT + NE;
    __bf16* xb   = attn + NE;
    __bf16* Wqb  = xb + NE;
    __bf16* Wkb  = Wqb + NW;
    __bf16* Wvb  = Wkb + NW;
    __bf16* Wob  = Wvb + NW;

    // One-shot f32 -> bf16 conversions (halves GEMM bytes, kills inner cvt).
    cvt_bf16_kernel<<<dim3((NE / 8) / 256), dim3(256), 0, stream>>>(x, xb, NE / 8);
    cvt_bf16_kernel<<<dim3((NW / 8) / 256), dim3(256), 0, stream>>>(Wq, Wqb, NW / 8);
    cvt_bf16_kernel<<<dim3((NW / 8) / 256), dim3(256), 0, stream>>>(Wk, Wkb, NW / 8);
    cvt_bf16_kernel<<<dim3((NW / 8) / 256), dim3(256), 0, stream>>>(Wv, Wvb, NW / 8);
    cvt_bf16_kernel<<<dim3((NW / 8) / 256), dim3(256), 0, stream>>>(Wo, Wob, NW / 8);

    dim3 g1(MTOT / 32, DD / 512, 3), b1(256);
    qkv_rope_kernel<<<g1, b1, 0, stream>>>(xb, fcos, fsin, Wqb, Wkb, Wvb,
                                           qbuf, kbuf, vT);

    dim3 g2(CC / 128, HH, BB), b2(256);
    attn_kernel<<<g2, b2, 0, stream>>>(qbuf, kbuf, vT, attn);

    dim3 g3(MTOT / 32, DD / 512), b3(256);
    out_proj_kernel<<<g3, b3, 0, stream>>>(attn, Wob, (float*)d_out);
}
